// LSTMModel_45329084842333
// MI455X (gfx1250) — matrix operand order repaired and checked by reading, not compiled, not run on hardware
//
#include <hip/hip_runtime.h>
#include <hip/hip_bf16.h>

typedef __attribute__((ext_vector_type(16))) _Float16 v16h;
typedef __attribute__((ext_vector_type(8)))  _Float16 v8h;
typedef __attribute__((ext_vector_type(8)))  float    v8f;

#define WMMA16(a, b, c) __builtin_amdgcn_wmma_f32_16x16x32_f16( \
    false, (a), false, (b), (short)0, (c), false, false)

static constexpr int BATCH = 512;
static constexpr int TSEQ  = 512;
static constexpr int DIN0  = 64;
static constexpr int NH1 = 64, NH2 = 128, NH3 = 64;
static constexpr int NF1 = 128, NF2 = 64, NCLS = 10, NCLSP = 16;

__device__ __forceinline__ v16h cat16(v8h lo, v8h hi) {
    return __builtin_shufflevector(lo, hi, 0,1,2,3,4,5,6,7,8,9,10,11,12,13,14,15);
}

// --- branchless, hardware-native activations (v_exp_f32 / v_rcp_f32 / v_tanh_f32)
__device__ __forceinline__ float fast_sigmoid(float x) {
    // 1 / (1 + 2^(-x*log2e))
    return __builtin_amdgcn_rcpf(1.0f + __builtin_amdgcn_exp2f(-1.4426950408889634f * x));
}
__device__ __forceinline__ float fast_tanh(float x) {
#if __has_builtin(__builtin_amdgcn_tanhf)
    return __builtin_amdgcn_tanhf(x);
#else
    // 2 / (1 + 2^(-2x*log2e)) - 1
    return 2.0f * __builtin_amdgcn_rcpf(
               1.0f + __builtin_amdgcn_exp2f(-2.8853900817779268f * x)) - 1.0f;
#endif
}

// ---------------------------------------------------------------------------
// LayerNorm over D=64; one wave per (b,t) row, 2 elems/lane; f16 output.
// ---------------------------------------------------------------------------
__global__ __launch_bounds__(256)
void ln_kernel(const float* __restrict__ x, const float* __restrict__ gamma,
               const float* __restrict__ beta, _Float16* __restrict__ out) {
    const int row  = blockIdx.x * 8 + (threadIdx.x >> 5);
    const int lane = threadIdx.x & 31;
    const float2 v = *(const float2*)(x + (size_t)row * 64 + lane * 2);
    float s = v.x + v.y;
    #pragma unroll
    for (int m = 16; m > 0; m >>= 1) s += __shfl_xor(s, m, 32);
    const float mu = s * (1.0f / 64.0f);
    const float dx = v.x - mu, dy = v.y - mu;
    float q = dx * dx + dy * dy;
    #pragma unroll
    for (int m = 16; m > 0; m >>= 1) q += __shfl_xor(q, m, 32);
    const float rstd = rsqrtf(q * (1.0f / 64.0f) + 1e-5f);
    _Float16* p = out + (size_t)row * 64 + lane * 2;
    p[0] = (_Float16)(dx * rstd * gamma[lane * 2 + 0] + beta[lane * 2 + 0]);
    p[1] = (_Float16)(dy * rstd * gamma[lane * 2 + 1] + beta[lane * 2 + 1]);
}

// ---------------------------------------------------------------------------
// Pack LSTM weights: wc[4H, Din+H] f16 = [w_ih | w_hh]; bs[4H] = b_ih + b_hh.
// ---------------------------------------------------------------------------
__global__ void pack_lstm(const float* __restrict__ w_ih, const float* __restrict__ w_hh,
                          const float* __restrict__ b_ih, const float* __restrict__ b_hh,
                          int Din, int H, _Float16* __restrict__ wc, float* __restrict__ bs) {
    const int KC = Din + H, G = 4 * H;
    const int idx = blockIdx.x * blockDim.x + threadIdx.x;
    if (idx < G * KC) {
        const int n = idx / KC, k = idx % KC;
        const float v = (k < Din) ? w_ih[n * Din + k] : w_hh[n * H + (k - Din)];
        wc[idx] = (_Float16)v;
    }
    if (idx < G) bs[idx] = b_ih[idx] + b_hh[idx];
}

// f32 -> f16 weight convert with row padding (pad rows zeroed).
__global__ void cvt_fc(const float* __restrict__ w, int N, int K, int Npad,
                       _Float16* __restrict__ out) {
    const int idx = blockIdx.x * blockDim.x + threadIdx.x;
    if (idx >= Npad * K) return;
    const int n = idx / K, k = idx % K;
    out[idx] = (_Float16)((n < N) ? w[n * K + k] : 0.0f);
}

// ---------------------------------------------------------------------------
// Fused LSTM layer. Each workgroup owns 16 batch rows for all T steps.
// Wave w owns H-columns [16w,16w+16): computes gate tiles i,f,g,o there,
// keeps c in a v8f register fragment, h double-buffered f16 in LDS.
// gates = [x_t | h] (16 x (DIN+H)) @ wc^T + b, wc row-major [4H, DIN+H].
// First WMMA of each chain uses inline-0 C; bias folded into the epilogue.
// ---------------------------------------------------------------------------
template <int DINL, int H, bool STORE_SEQ>
__global__ __launch_bounds__(32 * (H / 16))
void lstm_kernel(const _Float16* __restrict__ xseq,   // [B, T, DINL] f16
                 const _Float16* __restrict__ wc,     // [4H, DINL+H] f16
                 const float*    __restrict__ bs,     // [4H] f32
                 _Float16*       __restrict__ outseq) // [B,T,H] or [B,H](last)
{
    constexpr int KC = DINL + H;
    __shared__ __align__(32) _Float16 hbuf[2][16][H];

    const int wv    = threadIdx.x >> 5;
    const int lane  = threadIdx.x & 31;
    const int l15   = lane & 15;
    const int hi    = lane >> 4;          // 0 or 1 (lane half)
    const int mbase = blockIdx.x * 16;

    for (int i = threadIdx.x; i < 16 * H; i += blockDim.x)
        hbuf[0][i / H][i % H] = (_Float16)0.0f;

    float bias[4];
    #pragma unroll
    for (int g = 0; g < 4; ++g) bias[g] = bs[g * H + wv * 16 + l15];

    v8f c = {};
    __syncthreads();

    const size_t rowT = (size_t)TSEQ * DINL;
    const _Float16* xrowb = xseq + (size_t)(mbase + l15) * rowT;
    int cur = 0;

    for (int t = 0; t < TSEQ; ++t) {
        v8f acc[4];
        const _Float16* xrow = xrowb + (size_t)t * DINL;

        // ---- peeled first K-step over x_t: inline-zero accumulator
        {
            v8h alo = *(const v8h*)(xrow + hi * 8);
            v8h ahi = *(const v8h*)(xrow + 16 + hi * 8);
            v16h a = cat16(alo, ahi);
            const v8f zero = {};
            #pragma unroll
            for (int g = 0; g < 4; ++g) {
                const _Float16* brow =
                    wc + (size_t)(g * H + wv * 16 + l15) * KC + hi * 16;
                v16h b = cat16(*(const v8h*)(brow - hi * 16 + hi * 8), *(const v8h*)(brow - hi * 16 + 16 + hi * 8));
                acc[g] = WMMA16(a, b, zero);
            }
        }
        // ---- remaining K over input x_t
        #pragma unroll
        for (int kk = 32; kk < DINL; kk += 32) {
            v8h alo = *(const v8h*)(xrow + kk + hi * 8);
            v8h ahi = *(const v8h*)(xrow + kk + 16 + hi * 8);
            v16h a = cat16(alo, ahi);
            #pragma unroll
            for (int g = 0; g < 4; ++g) {
                const _Float16* brow =
                    wc + (size_t)(g * H + wv * 16 + l15) * KC + kk + hi * 16;
                v16h b = cat16(*(const v8h*)(brow - hi * 16 + hi * 8), *(const v8h*)(brow - hi * 16 + 16 + hi * 8));
                acc[g] = WMMA16(a, b, acc[g]);
            }
        }
        // ---- K over recurrent h (from LDS)
        #pragma unroll
        for (int kk = 0; kk < H; kk += 32) {
            const _Float16* hrow = &hbuf[cur][l15][0];
            v8h alo = *(const v8h*)(hrow + kk + hi * 8);
            v8h ahi = *(const v8h*)(hrow + kk + 16 + hi * 8);
            v16h a = cat16(alo, ahi);
            #pragma unroll
            for (int g = 0; g < 4; ++g) {
                const _Float16* brow =
                    wc + (size_t)(g * H + wv * 16 + l15) * KC + DINL + kk + hi * 16;
                v16h b = cat16(*(const v8h*)(brow - hi * 16 + hi * 8), *(const v8h*)(brow - hi * 16 + 16 + hi * 8));
                acc[g] = WMMA16(a, b, acc[g]);
            }
        }

        // ---- branchless activations + cell update, all in registers
        const int col = wv * 16 + l15;
        #pragma unroll
        for (int e = 0; e < 8; ++e) {
            const float ig = fast_sigmoid(acc[0][e] + bias[0]);
            const float fg = fast_sigmoid(acc[1][e] + bias[1]);
            const float gg = fast_tanh(acc[2][e] + bias[2]);
            const float og = fast_sigmoid(acc[3][e] + bias[3]);
            const float cn = fg * c[e] + ig * gg;
            c[e] = cn;
            const float hv = og * fast_tanh(cn);
            const int m = e + hi * 8;
            hbuf[cur ^ 1][m][col] = (_Float16)hv;  // pre-ReLU for recurrence
            const float hr = hv > 0.0f ? hv : 0.0f;
            if (STORE_SEQ) {
                outseq[((size_t)(mbase + m) * TSEQ + t) * H + col] = (_Float16)hr;
            } else if (t == TSEQ - 1) {
                outseq[(size_t)(mbase + m) * H + col] = (_Float16)hr;
            }
        }
        __syncthreads();
        cur ^= 1;
    }
}

// ---------------------------------------------------------------------------
// WMMA GEMM: out[M,Npad] = act(X[M,K] @ W[Npad,K]^T + bias). One wave/tile.
// ---------------------------------------------------------------------------
template <bool RELU, bool OUTF16>
__global__ __launch_bounds__(32)
void fc_kernel(const _Float16* __restrict__ X, const _Float16* __restrict__ W,
               const float* __restrict__ bias, int Npad, int Nreal, int K,
               void* __restrict__ out) {
    const int lane = threadIdx.x & 31;
    const int l15 = lane & 15, hi = lane >> 4;
    const int mbase = blockIdx.x * 16, nbase = blockIdx.y * 16;
    const _Float16* xrow = X + (size_t)(mbase + l15) * K;
    const _Float16* wrow = W + (size_t)(nbase + l15) * K;

    v8f acc;
    {   // peeled first K-step with inline-zero C
        v8h alo = *(const v8h*)(xrow + hi * 8);
        v8h ahi = *(const v8h*)(xrow + 16 + hi * 8);
        v16h a = cat16(alo, ahi);
        v16h b = cat16(*(const v8h*)(wrow + hi * 8), *(const v8h*)(wrow + 16 + hi * 8));
        const v8f zero = {};
        acc = WMMA16(a, b, zero);
    }
    for (int kk = 32; kk < K; kk += 32) {
        v8h alo = *(const v8h*)(xrow + kk + hi * 8);
        v8h ahi = *(const v8h*)(xrow + kk + 16 + hi * 8);
        v16h a = cat16(alo, ahi);
        v16h b = cat16(*(const v8h*)(wrow + kk + hi * 8), *(const v8h*)(wrow + kk + 16 + hi * 8));
        acc = WMMA16(a, b, acc);
    }
    const int col = nbase + l15;
    const float bb = (col < Nreal) ? bias[col] : 0.0f;
    #pragma unroll
    for (int e = 0; e < 8; ++e) {
        float v = acc[e] + bb;
        if (RELU) v = v > 0.0f ? v : 0.0f;
        const int row = mbase + e + hi * 8;
        if (OUTF16) ((_Float16*)out)[(size_t)row * Npad + col] = (_Float16)v;
        else        ((float*)out)[(size_t)row * Npad + col] = v;
    }
}

__global__ __launch_bounds__(256)
void softmax_kernel(const float* __restrict__ logits, float* __restrict__ out) {
    const int r = blockIdx.x * blockDim.x + threadIdx.x;
    if (r >= BATCH) return;
    const float* p = logits + (size_t)r * NCLSP;
    float mx = p[0];
    #pragma unroll
    for (int i = 1; i < NCLS; ++i) mx = fmaxf(mx, p[i]);
    float e[NCLS], s = 0.0f;
    #pragma unroll
    for (int i = 0; i < NCLS; ++i) {
        e[i] = __builtin_amdgcn_exp2f(1.4426950408889634f * (p[i] - mx));
        s += e[i];
    }
    const float inv = 1.0f / s;   // full-rate divide acceptable here; exact recip
    #pragma unroll
    for (int i = 0; i < NCLS; ++i) out[(size_t)r * NCLS + i] = e[i] * inv;
}

// ---------------------------------------------------------------------------
extern "C" void kernel_launch(void* const* d_in, const int* in_sizes, int n_in,
                              void* d_out, int out_size, void* d_ws, size_t ws_size,
                              hipStream_t stream) {
    const float* x     = (const float*)d_in[0];
    const float* gamma = (const float*)d_in[1];
    const float* beta  = (const float*)d_in[2];
    const float* w_ih1 = (const float*)d_in[3];
    const float* w_hh1 = (const float*)d_in[4];
    const float* b_ih1 = (const float*)d_in[5];
    const float* b_hh1 = (const float*)d_in[6];
    const float* w_ih2 = (const float*)d_in[7];
    const float* w_hh2 = (const float*)d_in[8];
    const float* b_ih2 = (const float*)d_in[9];
    const float* b_hh2 = (const float*)d_in[10];
    const float* w_ih3 = (const float*)d_in[11];
    const float* w_hh3 = (const float*)d_in[12];
    const float* b_ih3 = (const float*)d_in[13];
    const float* b_hh3 = (const float*)d_in[14];
    const float* fc1_w = (const float*)d_in[15];
    const float* fc1_b = (const float*)d_in[16];
    const float* fc2_w = (const float*)d_in[17];
    const float* fc2_b = (const float*)d_in[18];
    const float* fc3_w = (const float*)d_in[19];
    const float* fc3_b = (const float*)d_in[20];

    uint8_t* wp = (uint8_t*)d_ws;
    auto carve = [&](size_t bytes) -> void* {
        void* p = wp;
        wp += (bytes + 255) & ~(size_t)255;
        return p;
    };
    _Float16* w1c  = (_Float16*)carve((size_t)4 * NH1 * (DIN0 + NH1) * 2);
    _Float16* w2c  = (_Float16*)carve((size_t)4 * NH2 * (NH1 + NH2) * 2);
    _Float16* w3c  = (_Float16*)carve((size_t)4 * NH3 * (NH2 + NH3) * 2);
    float*    b1s  = (float*)carve((size_t)4 * NH1 * 4);
    float*    b2s  = (float*)carve((size_t)4 * NH2 * 4);
    float*    b3s  = (float*)carve((size_t)4 * NH3 * 4);
    _Float16* f1h  = (_Float16*)carve((size_t)NF1 * NH3 * 2);
    _Float16* f2h  = (_Float16*)carve((size_t)NF2 * NF1 * 2);
    _Float16* f3h  = (_Float16*)carve((size_t)NCLSP * NF2 * 2);
    _Float16* lnx  = (_Float16*)carve((size_t)BATCH * TSEQ * DIN0 * 2);
    _Float16* h1s  = (_Float16*)carve((size_t)BATCH * TSEQ * NH1 * 2);
    _Float16* h2s  = (_Float16*)carve((size_t)BATCH * TSEQ * NH2 * 2);
    _Float16* h3l  = (_Float16*)carve((size_t)BATCH * NH3 * 2);
    _Float16* a1   = (_Float16*)carve((size_t)BATCH * NF1 * 2);
    _Float16* a2   = (_Float16*)carve((size_t)BATCH * NF2 * 2);
    float*    lg   = (float*)carve((size_t)BATCH * NCLSP * 4);

    // weight packing / conversion
    {
        int t1 = 4 * NH1 * (DIN0 + NH1);
        pack_lstm<<<(t1 + 255) / 256, 256, 0, stream>>>(w_ih1, w_hh1, b_ih1, b_hh1, DIN0, NH1, w1c, b1s);
        int t2 = 4 * NH2 * (NH1 + NH2);
        pack_lstm<<<(t2 + 255) / 256, 256, 0, stream>>>(w_ih2, w_hh2, b_ih2, b_hh2, NH1, NH2, w2c, b2s);
        int t3 = 4 * NH3 * (NH2 + NH3);
        pack_lstm<<<(t3 + 255) / 256, 256, 0, stream>>>(w_ih3, w_hh3, b_ih3, b_hh3, NH2, NH3, w3c, b3s);
        cvt_fc<<<(NF1 * NH3 + 255) / 256, 256, 0, stream>>>(fc1_w, NF1, NH3, NF1, f1h);
        cvt_fc<<<(NF2 * NF1 + 255) / 256, 256, 0, stream>>>(fc2_w, NF2, NF1, NF2, f2h);
        cvt_fc<<<(NCLSP * NF2 + 255) / 256, 256, 0, stream>>>(fc3_w, NCLS, NF2, NCLSP, f3h);
    }

    // LayerNorm: 262144 rows, 8 rows/block
    ln_kernel<<<(BATCH * TSEQ) / 8, 256, 0, stream>>>(x, gamma, beta, lnx);

    // LSTM stack: 32 workgroups of 16 batch rows each
    lstm_kernel<DIN0, NH1, true ><<<BATCH / 16, 32 * (NH1 / 16), 0, stream>>>(lnx, w1c, b1s, h1s);
    lstm_kernel<NH1,  NH2, true ><<<BATCH / 16, 32 * (NH2 / 16), 0, stream>>>(h1s, w2c, b2s, h2s);
    lstm_kernel<NH2,  NH3, false><<<BATCH / 16, 32 * (NH3 / 16), 0, stream>>>(h2s, w3c, b3s, h3l);

    // MLP head
    fc_kernel<true,  true ><<<dim3(BATCH / 16, NF1 / 16),  32, 0, stream>>>(h3l, f1h, fc1_b, NF1,  NF1,  NH3, a1);
    fc_kernel<true,  true ><<<dim3(BATCH / 16, NF2 / 16),  32, 0, stream>>>(a1,  f2h, fc2_b, NF2,  NF2,  NF1, a2);
    fc_kernel<false, false><<<dim3(BATCH / 16, NCLSP / 16), 32, 0, stream>>>(a2, f3h, fc3_b, NCLSP, NCLS, NF2, lg);

    softmax_kernel<<<(BATCH + 255) / 256, 256, 0, stream>>>(lg, (float*)d_out);
}
